// rnn_decoder_14173392077510
// MI455X (gfx1250) — matrix operand order repaired and checked by reading, not compiled, not run on hardware
//
#include <hip/hip_runtime.h>
#include <hip/hip_bf16.h>
#include <math.h>

// ---------------- problem constants ----------------
#define TT 128
#define BB 128
#define SS 128
#define HH 1024
#define EE 512
#define LL 2
#define GG 4096   // 4*H

typedef __attribute__((ext_vector_type(16))) __bf16 v16bf;
typedef __attribute__((ext_vector_type(8)))  __bf16 v8bf;
typedef __attribute__((ext_vector_type(8)))  float  v8f;

__device__ __forceinline__ float sigmoidf_(float x) { return 1.0f / (1.0f + expf(-x)); }

// ---------------- conversion / init kernels ----------------
__global__ void cvt_bf16_kernel(const float* __restrict__ src, __bf16* __restrict__ dst, int n) {
    int i = blockIdx.x * blockDim.x + threadIdx.x;
    if (i < n) dst[i] = (__bf16)src[i];
}

__global__ void cvt_bf16_strided_kernel(const float* __restrict__ src, __bf16* __restrict__ dst,
                                        int rows, int cols, int dld) {
    int i = blockIdx.x * blockDim.x + threadIdx.x;
    if (i < rows * cols) {
        int r = i / cols, c = i % cols;
        dst[(size_t)r * dld + c] = (__bf16)src[i];
    }
}

__global__ void copy_f32_kernel(const float* __restrict__ src, float* __restrict__ dst, int n) {
    int i = blockIdx.x * blockDim.x + threadIdx.x;
    if (i < n) dst[i] = src[i];
}

// embedding gather + fp32->bf16 : one block per (t,b) row of E=512
__global__ void gather_embed_kernel(const int* __restrict__ tokens, const float* __restrict__ emb,
                                    __bf16* __restrict__ out) {
    int row = blockIdx.x;                 // T*B rows
    int tok = tokens[row];
    const float* src = emb + (size_t)tok * EE;
    __bf16* dst = out + (size_t)row * EE;
    for (int i = threadIdx.x; i < EE; i += blockDim.x) dst[i] = (__bf16)src[i];
}

// ---------------- shared WMMA K-loop helper ----------------
// Accumulates 4 independent 16x16 output tiles that share the same A tile.
// Wn[j] = row-base pointer (per-lane, pre-offset by half*16) of the j-th weight row group.
// A layout (16-bit, 16x32 MxK): lane -> M=lane%16; per-lane K chunks [16*half..+7], [16*half+8..+7]
// B layout (32x16 KxN): lane%16 -> N (row of W[N,K]); half -> contiguous K [16*half..+15]
__device__ __forceinline__ void wmma_kloop4(const __bf16* __restrict__ ap, int a_off0, int a_off1,
                                            const __bf16* __restrict__ wp0, const __bf16* __restrict__ wp1,
                                            const __bf16* __restrict__ wp2, const __bf16* __restrict__ wp3,
                                            int K, v8f* acc) {
    for (int k = 0; k < K; k += 32) {
        union { v16bf v; v8bf h[2]; } au;
        au.h[0] = *(const v8bf*)(ap + k + a_off0);
        au.h[1] = *(const v8bf*)(ap + k + a_off1);
        v16bf a  = au.v;
        v16bf b0 = *(const v16bf*)(wp0 + k);
        v16bf b1 = *(const v16bf*)(wp1 + k);
        v16bf b2 = *(const v16bf*)(wp2 + k);
        v16bf b3 = *(const v16bf*)(wp3 + k);
        acc[0] = __builtin_amdgcn_wmma_f32_16x16x32_bf16(false, a, false, b0, (short)0, acc[0], false, false);
        acc[1] = __builtin_amdgcn_wmma_f32_16x16x32_bf16(false, a, false, b1, (short)0, acc[1], false, false);
        acc[2] = __builtin_amdgcn_wmma_f32_16x16x32_bf16(false, a, false, b2, (short)0, acc[2], false, false);
        acc[3] = __builtin_amdgcn_wmma_f32_16x16x32_bf16(false, a, false, b3, (short)0, acc[3], false, false);
    }
}

// ---------------- fused LSTM layer: dual GEMM + cell epilogue ----------------
// Each wave computes the SAME 16(M) x 16(N) hidden tile for ALL FOUR gates:
//   acc[g] = (x * Wih^T + h * Whh^T)[m, g*H + n],  g in {i,f,g,o}
// Epilogue applies biases + LSTM cell in-register: no gates buffer, no separate cell kernel.
// grid = (H/16, B/16), block = 32 (one wave).
__global__ __launch_bounds__(32)
void lstm_fused_kernel(const __bf16* __restrict__ X,  int ldx, const __bf16* __restrict__ Wih, int Kx,
                       const __bf16* __restrict__ Hp, int ldh, const __bf16* __restrict__ Whh,
                       const float* __restrict__ bih, const float* __restrict__ bhh,
                       float* __restrict__ h, float* __restrict__ c,
                       __bf16* __restrict__ hbf, int hbf_ld) {
    const int lane = threadIdx.x & 31;
    const int half = lane >> 4;
    const int lrow = lane & 15;
    const int n_base = blockIdx.x * 16;        // hidden-unit tile
    const int m_base = blockIdx.y * 16;        // batch tile
    const int a_off0 = 16 * half;
    const int a_off1 = 16 * half + 8;

    v8f acc[4] = {};

    // x * Wih^T  (W rows: g*H + n)
    {
        const __bf16* ap  = X + (size_t)(m_base + lrow) * ldx;
        const __bf16* w0 = Wih + (size_t)(0 * HH + n_base + lrow) * Kx + half * 16;
        const __bf16* w1 = Wih + (size_t)(1 * HH + n_base + lrow) * Kx + half * 16;
        const __bf16* w2 = Wih + (size_t)(2 * HH + n_base + lrow) * Kx + half * 16;
        const __bf16* w3 = Wih + (size_t)(3 * HH + n_base + lrow) * Kx + half * 16;
        wmma_kloop4(ap, a_off0, a_off1, w0, w1, w2, w3, Kx, acc);
    }
    // h_prev * Whh^T  (K = H always)
    {
        const __bf16* ap  = Hp + (size_t)(m_base + lrow) * ldh;
        const __bf16* w0 = Whh + (size_t)(0 * HH + n_base + lrow) * HH + half * 16;
        const __bf16* w1 = Whh + (size_t)(1 * HH + n_base + lrow) * HH + half * 16;
        const __bf16* w2 = Whh + (size_t)(2 * HH + n_base + lrow) * HH + half * 16;
        const __bf16* w3 = Whh + (size_t)(3 * HH + n_base + lrow) * HH + half * 16;
        wmma_kloop4(ap, a_off0, a_off1, w0, w1, w2, w3, HH, acc);
    }

    // LSTM cell epilogue; D layout: vgpr r, lane -> m = m_base + 8*half + r, n = n_base + lrow
    const int n = n_base + lrow;
    const float bi = bih[n]          + bhh[n];
    const float bf = bih[HH + n]     + bhh[HH + n];
    const float bg = bih[2 * HH + n] + bhh[2 * HH + n];
    const float bo = bih[3 * HH + n] + bhh[3 * HH + n];

#pragma unroll
    for (int r = 0; r < 8; ++r) {
        const int m = m_base + (half << 3) + r;
        const size_t idx = (size_t)m * HH + n;
        float gi = acc[0][r] + bi;
        float gf = acc[1][r] + bf;
        float gg = acc[2][r] + bg;
        float go = acc[3][r] + bo;
        float cn = sigmoidf_(gf) * c[idx] + sigmoidf_(gi) * tanhf(gg);
        float hn = sigmoidf_(go) * tanhf(cn);
        c[idx] = cn;
        h[idx] = hn;
        hbf[(size_t)m * hbf_ld + n] = (__bf16)hn;
    }
}

// ---------------- generic WMMA GEMM (+bias, +tanh) ----------------
// D[128,N] = A[128,K](lda) * W[N,K]^T; one wave computes 16(M) x 64(N).
// grid = (N/64, 128/16), block = 32.
__global__ __launch_bounds__(32)
void gemm_bf16_wmma_kernel(const __bf16* __restrict__ A, int lda, const __bf16* __restrict__ W, int K,
                           const float* __restrict__ bias, float* __restrict__ D, int N, int act) {
    const int lane = threadIdx.x & 31;
    const int half = lane >> 4;
    const int lrow = lane & 15;
    const int n_base = blockIdx.x * 64;
    const int m_base = blockIdx.y * 16;
    const int a_off0 = 16 * half;
    const int a_off1 = 16 * half + 8;

    v8f acc[4] = {};
    const __bf16* ap = A + (size_t)(m_base + lrow) * lda;
    const __bf16* w0 = W + (size_t)(n_base +  0 + lrow) * K + half * 16;
    const __bf16* w1 = W + (size_t)(n_base + 16 + lrow) * K + half * 16;
    const __bf16* w2 = W + (size_t)(n_base + 32 + lrow) * K + half * 16;
    const __bf16* w3 = W + (size_t)(n_base + 48 + lrow) * K + half * 16;
    wmma_kloop4(ap, a_off0, a_off1, w0, w1, w2, w3, K, acc);

#pragma unroll
    for (int j = 0; j < 4; ++j) {
        const int n = n_base + j * 16 + lrow;
        const float bv = bias ? bias[n] : 0.0f;
#pragma unroll
        for (int r = 0; r < 8; ++r) {
            const int m = m_base + (half << 3) + r;
            float v = acc[j][r] + bv;
            if (act) v = tanhf(v);
            D[(size_t)m * N + n] = v;
        }
    }
}

// ---------------- attention scores + softmax + context ----------------
// one block (128 threads) per batch element; writes bf16 c_t into concat[:,0:H] (ld = ct_ld)
__global__ void attn_kernel(const float* __restrict__ contexts, const float* __restrict__ gamma,
                            __bf16* __restrict__ ct_out, int ct_ld) {
    const int b = blockIdx.x;
    const int tid = threadIdx.x;             // 0..127, one score per thread
    __shared__ float g[HH];
    __shared__ float w[SS];
    __shared__ float red[SS];

    for (int i = tid; i < HH; i += SS) g[i] = gamma[(size_t)b * HH + i];
    __syncthreads();

    const float* ctx = contexts + ((size_t)b * SS + tid) * HH;
    float s = 0.0f;
    for (int k = 0; k < HH; ++k) s += ctx[k] * g[k];

    red[tid] = s; __syncthreads();
    for (int off = SS / 2; off > 0; off >>= 1) {
        if (tid < off) red[tid] = fmaxf(red[tid], red[tid + off]);
        __syncthreads();
    }
    const float mx = red[0]; __syncthreads();

    float e = expf(s - mx);
    red[tid] = e; __syncthreads();
    for (int off = SS / 2; off > 0; off >>= 1) {
        if (tid < off) red[tid] += red[tid + off];
        __syncthreads();
    }
    w[tid] = e / red[0];
    __syncthreads();

    const float* cb = contexts + (size_t)b * SS * HH;
    for (int n = tid; n < HH; n += SS) {
        float acc = 0.0f;
        for (int ss = 0; ss < SS; ++ss) acc += w[ss] * cb[(size_t)ss * HH + n];
        ct_out[(size_t)b * ct_ld + n] = (__bf16)acc;
    }
}

// ---------------- host launcher ----------------
extern "C" void kernel_launch(void* const* d_in, const int* in_sizes, int n_in,
                              void* d_out, int out_size, void* d_ws, size_t ws_size,
                              hipStream_t stream) {
    (void)in_sizes; (void)n_in; (void)out_size; (void)ws_size;
    const int*   tokens    = (const int*)d_in[0];
    const float* h0        = (const float*)d_in[1];
    const float* c0        = (const float*)d_in[2];
    const float* contexts  = (const float*)d_in[3];
    const float* emb       = (const float*)d_in[4];
    const float* W_ih0     = (const float*)d_in[5];
    const float* W_hh0     = (const float*)d_in[6];
    const float* b_ih0     = (const float*)d_in[7];
    const float* b_hh0     = (const float*)d_in[8];
    const float* W_ih1     = (const float*)d_in[9];
    const float* W_hh1     = (const float*)d_in[10];
    const float* b_ih1     = (const float*)d_in[11];
    const float* b_hh1     = (const float*)d_in[12];
    const float* att_in_W  = (const float*)d_in[13];
    const float* att_in_b  = (const float*)d_in[14];
    const float* att_out_W = (const float*)d_in[15];
    const float* att_out_b = (const float*)d_in[16];

    // workspace carve-out (bf16 weights stay L2-resident: ~35 MB total)
    char* base = (char*)d_ws;
    size_t off = 0;
    auto carve = [&](size_t bytes) -> void* {
        void* p = base + off;
        off = (off + bytes + 255) & ~(size_t)255;
        return p;
    };
    __bf16* embs_bf  = (__bf16*)carve((size_t)TT * BB * EE * 2);
    __bf16* Wih0_bf  = (__bf16*)carve((size_t)GG * EE * 2);
    __bf16* Whh0_bf  = (__bf16*)carve((size_t)GG * HH * 2);
    __bf16* Wih1_bf  = (__bf16*)carve((size_t)GG * HH * 2);
    __bf16* Whh1_bf  = (__bf16*)carve((size_t)GG * HH * 2);
    __bf16* AinW_bf  = (__bf16*)carve((size_t)HH * HH * 2);
    __bf16* AoutW_bf = (__bf16*)carve((size_t)HH * 2 * HH * 2);
    float*  gamma    = (float*)carve((size_t)BB * HH * 4);
    __bf16* h0bf     = (__bf16*)carve((size_t)BB * HH * 2);       // layer-0 h state (bf16)
    __bf16* concat   = (__bf16*)carve((size_t)BB * 2 * HH * 2);   // [c_t | h_l1] per row, ld=2H

    float* out    = (float*)d_out;
    float* hstate = out + (size_t)TT * BB * HH;           // [L,B,H], ends as hT
    float* cstate = hstate + (size_t)LL * BB * HH;        // [L,B,H], ends as cT

    auto cvt = [&](const float* s, __bf16* d, size_t n) {
        cvt_bf16_kernel<<<dim3((unsigned)((n + 255) / 256)), dim3(256), 0, stream>>>(s, d, (int)n);
    };

    // one-time prep
    cvt(W_ih0,     Wih0_bf,  (size_t)GG * EE);
    cvt(W_hh0,     Whh0_bf,  (size_t)GG * HH);
    cvt(W_ih1,     Wih1_bf,  (size_t)GG * HH);
    cvt(W_hh1,     Whh1_bf,  (size_t)GG * HH);
    cvt(att_in_W,  AinW_bf,  (size_t)HH * HH);
    cvt(att_out_W, AoutW_bf, (size_t)HH * 2 * HH);
    gather_embed_kernel<<<dim3(TT * BB), dim3(128), 0, stream>>>(tokens, emb, embs_bf);
    copy_f32_kernel<<<dim3(LL * BB * HH / 256), dim3(256), 0, stream>>>(h0, hstate, LL * BB * HH);
    copy_f32_kernel<<<dim3(LL * BB * HH / 256), dim3(256), 0, stream>>>(c0, cstate, LL * BB * HH);
    cvt(h0, h0bf, (size_t)BB * HH);                                       // layer-0 h -> bf16
    cvt_bf16_strided_kernel<<<dim3(BB * HH / 256), dim3(256), 0, stream>>>(
        h0 + (size_t)BB * HH, concat + HH, BB, HH, 2 * HH);               // layer-1 h -> concat[:,H:]

    const dim3 blk(32);
    const dim3 grid_l(HH / 16, BB / 16);   // fused LSTM: 64 x 8 waves
    const dim3 grid_h(HH / 64, BB / 16);   // H-wide GEMMs (N=1024)

    for (int t = 0; t < TT; ++t) {
        // layer 0: fused (x_t*Wih0^T + h*Whh0^T) + cell; updates hstate/cstate[0], h0bf
        lstm_fused_kernel<<<grid_l, blk, 0, stream>>>(
            embs_bf + (size_t)t * BB * EE, EE, Wih0_bf, EE,
            h0bf, HH, Whh0_bf,
            b_ih0, b_hh0, hstate, cstate, h0bf, HH);

        // layer 1: fused; h state (bf16) lives in concat[:,H:] (ld 2H)
        lstm_fused_kernel<<<grid_l, blk, 0, stream>>>(
            h0bf, HH, Wih1_bf, HH,
            concat + HH, 2 * HH, Whh1_bf,
            b_ih1, b_hh1, hstate + (size_t)BB * HH, cstate + (size_t)BB * HH,
            concat + HH, 2 * HH);

        // attention: gamma = h_l1 * att_in_W^T + b
        gemm_bf16_wmma_kernel<<<grid_h, blk, 0, stream>>>(
            concat + HH, 2 * HH, AinW_bf, HH,
            att_in_b, gamma, HH, 0);
        attn_kernel<<<dim3(BB), dim3(SS), 0, stream>>>(contexts, gamma, concat, 2 * HH);

        // out_t = tanh([c_t, h_l1] * att_out_W^T + b)
        gemm_bf16_wmma_kernel<<<grid_h, blk, 0, stream>>>(
            concat, 2 * HH, AoutW_bf, 2 * HH,
            att_out_b, out + (size_t)t * BB * HH, HH, 1);
    }
}
